// CoLA_3959959847200
// MI455X (gfx1250) — matrix operand order repaired and checked by reading, not compiled, not run on hardware
//
#include <hip/hip_runtime.h>
#include <math.h>
#include <stdint.h>

// ---------------------------------------------------------------------------
// CoLA forward on MI455X (gfx1250, wave32, WMMA).
// All heavy math is routed through v_wmma_f32_16x16x32_bf16:
//   embed conv (M=12000,K=6144,N=2048)  ~302 GFLOP
//   cas        (M=12000,K=2048,N=20)
//   QKV hoisted over 10 unique chunks (M=5920,K=2048,N=6144) ~149 GFLOP
//   out-proj batched over 24 MHA calls (M=14208,K=2048,N=2048) ~119 GFLOP
//   rh conv1/conv2 as padded-window GEMMs (K=6144 / K=384)
// ~600 GFLOP vs ~0.5GB traffic -> compute bound -> WMMA path.
// A-tiles stream global->LDS via GLOBAL_LOAD_ASYNC_TO_LDS_B128 (ASYNCcnt),
// double-buffered so the async engine overlaps WMMA issue.
// ---------------------------------------------------------------------------

typedef __bf16 bf16_t;
typedef __attribute__((ext_vector_type(16))) __bf16 v16bf;
typedef __attribute__((ext_vector_type(8)))  float  v8f;

#define NB 16      // batch
#define NT 750     // time
#define NC 2048    // channels

// ---------------------------------------------------------------------------
// Generic bf16 WMMA GEMM: C = act(A*B + bias), A rows addressed in "batched
// window" form (rows_per_batch / batch_stride / lda) so conv1d-with-pad
// becomes a plain GEMM over overlapping rows.  B is [K][N] bf16.
// ---------------------------------------------------------------------------
struct GemmP {
  const bf16_t* A; long a_rpb, a_bstride, a_lda;
  const bf16_t* Bm;
  const float* bias;
  float* outF; bf16_t* outB;
  long o_rpb, o_bstride, o_ld;
  int M, N, K, act;   // act: 0 none, 1 relu, 2 leaky(0.2)
};

#define BLKT 128
#define BKK  32

__global__ __launch_bounds__(256)
void k_gemm_wmma(GemmP p) {
  __shared__ bf16_t As[2][BLKT * BKK];   // [m][k] row-major, k contiguous
  __shared__ bf16_t Bs[2][BLKT * BKK];   // [n][k] (transposed in LDS)
  const int tid   = threadIdx.x;
  const int wave  = tid >> 5;
  const int lane  = tid & 31;
  const int hf    = lane >> 4;        // lane half (A/B/C layout split)
  const int l15   = lane & 15;
  const int waveM = wave & 3;         // 4 waves along M
  const int waveN = wave >> 2;        // 2 waves along N
  const long m0 = (long)blockIdx.y * BLKT;
  const long n0 = (long)blockIdx.x * BLKT;

  v8f acc[2][4];
  #pragma unroll
  for (int i = 0; i < 2; ++i)
    #pragma unroll
    for (int j = 0; j < 4; ++j)
      #pragma unroll
      for (int r = 0; r < 8; ++r) acc[i][j][r] = 0.0f;

  // ---- staging: A tile via async global->LDS (tracked by ASYNCcnt) --------
  const int ar   = tid >> 1;           // 0..127 : A row within tile
  const int aseg = tid & 1;            // two 16-elem (32B) segments
  const long agrow = m0 + ar;
  const bf16_t* a_rowp = nullptr;
  if (agrow < p.M) {
    long ab = agrow / p.a_rpb, arr = agrow % p.a_rpb;
    a_rowp = p.A + ab * p.a_bstride + arr * p.a_lda + aseg * 16;
  }
  // ---- staging: B tile, two K-rows packed into dwords ---------------------
  const int bkp  = tid & 15;           // K pair-of-rows index (0..15 -> k=2*bkp)
  const int bseg = tid >> 4;           // 0..15 : 8 consecutive n each
  const bool bFast = (p.N % 8 == 0) && (n0 + BLKT <= (long)p.N);

  auto stage = [&](int buf, int k0) {
    // A: 32 bytes per thread, contiguous in both global and LDS.
    bf16_t* adst = &As[buf][ar * BKK + aseg * 16];
    if (a_rowp) {
      unsigned lds = (unsigned)(uintptr_t)adst;          // low 32 bits = LDS addr
      unsigned long long ga = (unsigned long long)(uintptr_t)(a_rowp + k0);
      asm volatile(
        "global_load_async_to_lds_b128 %0, %1, off\n\t"
        "global_load_async_to_lds_b128 %0, %1, off offset:16"
        :: "v"(lds), "v"(ga) : "memory");
    } else {
      unsigned* d32 = (unsigned*)adst;
      #pragma unroll
      for (int e = 0; e < 8; ++e) d32[e] = 0u;
    }
    // B: pack (k,k+1) pairs per n so LDS sees dword stores and the fragment
    // loads stay 128-bit.
    unsigned* b32 = (unsigned*)&Bs[buf][0];
    if (bFast) {
      const bf16_t* r0 = p.Bm + (long)(k0 + 2 * bkp) * p.N + n0 + bseg * 8;
      const bf16_t* r1 = r0 + p.N;
      if (k0 + BKK < p.K) __builtin_prefetch(r0 + (long)BKK * p.N, 0, 1);
      uint4 u0 = *(const uint4*)r0;
      uint4 u1 = *(const uint4*)r1;
      unsigned w0[4] = {u0.x, u0.y, u0.z, u0.w};
      unsigned w1[4] = {u1.x, u1.y, u1.z, u1.w};
      #pragma unroll
      for (int i = 0; i < 4; ++i) {
        int nA = bseg * 8 + 2 * i, nBn = nA + 1;
        b32[nA  * (BKK / 2) + bkp] = (w0[i] & 0xffffu) | (w1[i] << 16);
        b32[nBn * (BKK / 2) + bkp] = (w0[i] >> 16) | (w1[i] & 0xffff0000u);
      }
    } else {
      // small-N fallback (N=20/32): bounds-checked scalar staging
      #pragma unroll
      for (int kk = 0; kk < 2; ++kk) {
        const bf16_t* src = p.Bm + (long)(k0 + 2 * bkp + kk) * p.N;
        #pragma unroll
        for (int i = 0; i < 8; ++i) {
          long col = n0 + bseg * 8 + i;
          bf16_t v = (col < p.N) ? src[col] : (bf16_t)0.0f;
          Bs[buf][(bseg * 8 + i) * BKK + 2 * bkp + kk] = v;
        }
      }
    }
  };

  const int nk = p.K / BKK;            // all K here are multiples of 32
  stage(0, 0);

  for (int kt = 0; kt < nk; ++kt) {
    const int cur = kt & 1;
    asm volatile("s_wait_asynccnt 0" ::: "memory");   // staging(cur) landed
    __syncthreads();
    if (kt + 1 < nk) stage(1 - cur, (kt + 1) * BKK);  // async prefetch next tile

    const unsigned* AsU = (const unsigned*)&As[cur][0];
    const unsigned* BsU = (const unsigned*)&Bs[cur][0];

    // --- fragments per ISA 16-bit layouts (A 16x32, B 32x16)
    union Fr { v16bf v; unsigned u[8]; };
    Fr afr[2], bfr[4];
    #pragma unroll
    for (int i = 0; i < 2; ++i) {
      int m = waveM * 32 + i * 16 + l15;
      const unsigned* row = AsU + m * (BKK / 2);
      #pragma unroll
      for (int j = 0; j < 4; ++j) afr[i].u[j] = row[j + 4 * hf];           // K = 2j    + 8*hf
      #pragma unroll
      for (int j = 4; j < 8; ++j) afr[i].u[j] = row[8 + (j - 4) + 4 * hf]; // K = 16+.. + 8*hf
    }
    #pragma unroll
    for (int jn = 0; jn < 4; ++jn) {
      int n = waveN * 64 + jn * 16 + l15;
      const unsigned* row = BsU + n * (BKK / 2);
      #pragma unroll
      for (int v = 0; v < 8; ++v) bfr[jn].u[v] = row[(v & 3) + 8 * (v >> 2) + 4 * hf]; // K = 2(v&3) + 16*(v>>2) + 8*hf
    }

    #pragma unroll
    for (int i = 0; i < 2; ++i)
      #pragma unroll
      for (int jn = 0; jn < 4; ++jn)
        acc[i][jn] = __builtin_amdgcn_wmma_f32_16x16x32_bf16(
            false, afr[i].v, false, bfr[jn].v, (short)0, acc[i][jn], false, false);
  }

  // --- epilogue: bias + activation, f32 and/or bf16 stores
  #pragma unroll
  for (int i = 0; i < 2; ++i)
    #pragma unroll
    for (int jn = 0; jn < 4; ++jn) {
      long n = n0 + waveN * 64 + jn * 16 + l15;
      if (n >= p.N) continue;
      float bv = p.bias ? p.bias[n] : 0.0f;
      #pragma unroll
      for (int r = 0; r < 8; ++r) {
        long m = m0 + waveM * 32 + i * 16 + r + 8 * hf;
        if (m >= p.M) continue;
        float v = acc[i][jn][r] + bv;
        if (p.act == 1)      v = v > 0.0f ? v : 0.0f;
        else if (p.act == 2) v = v >= 0.0f ? v : 0.2f * v;
        long ob = m / p.o_rpb, orr = m % p.o_rpb;
        long off = ob * p.o_bstride + orr * p.o_ld + n;
        if (p.outF) p.outF[off] = v;
        if (p.outB) p.outB[off] = (bf16_t)v;
      }
    }
}

// ---------------------------------------------------------------------------
// Small helper kernels
// ---------------------------------------------------------------------------
__global__ void k_fill_zero(unsigned* pp, long n) {
  long i = (long)blockIdx.x * blockDim.x + threadIdx.x;
  long st = (long)gridDim.x * blockDim.x;
  for (; i < n; i += st) pp[i] = 0u;
}

// x (B,T,C) f32 -> zero-padded bf16 rows [B][T+2][C] (rows 1..T)
__global__ void k_x_to_xpad(const float* x, bf16_t* xpad) {
  long n = (long)NB * NT * NC;
  long i = (long)blockIdx.x * blockDim.x + threadIdx.x;
  long st = (long)gridDim.x * blockDim.x;
  for (; i < n; i += st) {
    int c = (int)(i % NC);
    long r = i / NC;
    int t = (int)(r % NT), b = (int)(r / NT);
    xpad[((long)(b * (NT + 2) + t + 1)) * NC + c] = (bf16_t)x[i];
  }
}

// weight (N,Ci,KW) f32  ->  bf16 [KW][Ci][N]   (KW=1 gives plain transpose)
__global__ void k_w_to_bf16(const float* w, bf16_t* dst, int N, int Ci, int KW) {
  long n = (long)N * Ci * KW;
  long i = (long)blockIdx.x * blockDim.x + threadIdx.x;
  long st = (long)gridDim.x * blockDim.x;
  for (; i < n; i += st) {
    int kw = (int)(i % KW);
    long r = i / KW;
    int ci = (int)(r % Ci), nn = (int)(r / Ci);
    dst[((long)kw * Ci + ci) * N + nn] = (bf16_t)w[i];
  }
}

__global__ void k_actionness(const float* cas, float* act) {
  int i = blockIdx.x * blockDim.x + threadIdx.x;
  if (i < NB * NT) {
    float s = 0.0f;
    for (int c = 0; c < 20; ++c) s += cas[(long)i * 20 + c];
    act[i] = s;
  }
}

__global__ void k_a2(const float* act, const float* w1, const float* b1,
                     const float* w2, const float* b2, float* out) {
  int b = blockIdx.x;
  __shared__ float h[NT];
  const float* a = act + (long)b * NT;
  for (int t = threadIdx.x; t < NT; t += blockDim.x) {
    float xm = t > 0 ? a[t - 1] : 0.0f, x0 = a[t], xp = t < NT - 1 ? a[t + 1] : 0.0f;
    float v = w1[0] * xm + w1[1] * x0 + w1[2] * xp + b1[0];
    h[t] = v > 0.0f ? v : 0.0f;
  }
  __syncthreads();
  for (int t = threadIdx.x; t < NT; t += blockDim.x) {
    float xm = t > 0 ? h[t - 1] : 0.0f, x0 = h[t], xp = t < NT - 1 ? h[t + 1] : 0.0f;
    float v = w2[0] * xm + w2[1] * x0 + w2[2] * xp + b2[0];
    out[(long)b * NT + t] = v > 0.0f ? v : 0.0f;
  }
}

__global__ void k_rev(const float* act, float* rev) {
  int b = blockIdx.x;
  __shared__ float red[256];
  const float* a = act + (long)b * NT;
  float m = -1e30f;
  for (int t = threadIdx.x; t < NT; t += 256) m = fmaxf(m, a[t]);
  red[threadIdx.x] = m;
  __syncthreads();
  for (int s = 128; s > 0; s >>= 1) {
    if ((int)threadIdx.x < s) red[threadIdx.x] = fmaxf(red[threadIdx.x], red[threadIdx.x + s]);
    __syncthreads();
  }
  float mx = red[0];
  for (int t = threadIdx.x; t < NT; t += 256) rev[(long)b * NT + t] = mx - a[t];
}

// Bitonic (score desc, tie -> lower index first, matching stable argsort(-s)).
// Optional outputs: top-k indices, median of 750, mean of top-k values.
__global__ __launch_bounds__(256)
void k_sort_select(const float* base, long bstride, long cstride, int ncols,
                   int stride, int n, int* idx_out, int k_idx,
                   float* median_out, float* mean_out, int k_mean) {
  int blk = blockIdx.x;
  const float* p = base + (long)(blk / ncols) * bstride + (long)(blk % ncols) * cstride;
  __shared__ float key[1024];
  __shared__ int   kid[1024];
  for (int s = 0; s < 4; ++s) {
    int i = threadIdx.x + s * 256;
    if (i < n) { key[i] = p[(long)i * stride]; kid[i] = i; }
    else       { key[i] = -3.4e38f;            kid[i] = 1 << 30; }
  }
  __syncthreads();
  for (unsigned k2 = 2; k2 <= 1024; k2 <<= 1)
    for (unsigned j = k2 >> 1; j > 0; j >>= 1) {
      for (int s = 0; s < 4; ++s) {
        unsigned i = threadIdx.x + s * 256u;
        unsigned x = i ^ j;
        if (x > i) {
          float ka = key[i], kb = key[x];
          int   ia = kid[i], ib = kid[x];
          bool aFirst = (ka > kb) || (ka == kb && ia < ib);
          bool desc = ((i & k2) == 0);
          if (desc ? !aFirst : aFirst) { key[i] = kb; key[x] = ka; kid[i] = ib; kid[x] = ia; }
        }
      }
      __syncthreads();
    }
  if (idx_out)
    for (int i = threadIdx.x; i < k_idx; i += 256) idx_out[(long)blk * k_idx + i] = kid[i];
  if (median_out && threadIdx.x == 0) median_out[blk] = 0.5f * (key[374] + key[375]);
  if (mean_out && threadIdx.x == 0) {
    float s = 0.0f;
    for (int i = 0; i < k_mean; ++i) s += key[i];
    mean_out[blk] = s / (float)k_mean;
  }
}

__global__ void k_morph(const float* act, const float* median, float* hs_act, float* hs_bkg) {
  int b = blockIdx.x;
  __shared__ float bin[NT];
  const float* a = act + (long)b * NT;
  float med = median[b];
  for (int t = threadIdx.x; t < NT; t += 256) bin[t] = a[t] > med ? 1.0f : 0.0f;
  __syncthreads();
  for (int t = threadIdx.x; t < NT; t += 256) {
    float e3 = 1.0f, e6 = 1.0f, d3 = 0.0f, d6 = 0.0f;
    for (int j = -1; j <= 1; ++j) {   // M=3 erosion & dilation window
      float v = (t + j >= 0 && t + j < NT) ? bin[t + j] : 0.0f;
      e3 = fminf(e3, v); d3 = fmaxf(d3, v);
    }
    for (int j = -3; j <= 2; ++j) {   // M=6 erosion
      float v = (t + j >= 0 && t + j < NT) ? bin[t + j] : 0.0f;
      e6 = fminf(e6, v);
    }
    for (int j = -2; j <= 3; ++j) {   // M=6 dilation
      float v = (t + j >= 0 && t + j < NT) ? bin[t + j] : 0.0f;
      d6 = fmaxf(d6, v);
    }
    hs_act[(long)b * NT + t] = a[t] * (e3 - e6);
    hs_bkg[(long)b * NT + t] = a[t] * (d6 - d3);
  }
}

// Gather selected embedding rows -> f32 output and (for first rows) bf16 chunks
__global__ void k_gather(const float* emb, const int* idx, int K, float* outF,
                         bf16_t* chunks, int cbase, int max_chunk_rows) {
  int blk = blockIdx.x;
  int b = blk / K, r = blk % K;
  int t = idx[(long)b * K + r];
  const float* src = emb + ((long)b * NT + t) * NC;
  float* dst = outF + ((long)b * K + r) * NC;
  bf16_t* cdst = nullptr;
  if (chunks && r < max_chunk_rows) {
    int ch = cbase + r / 37, rr = r % 37;
    cdst = chunks + ((long)(ch * 16 + b) * 37 + rr) * NC;
  }
  for (int c = threadIdx.x; c < NC; c += 256) {
    float v = src[c];
    dst[c] = v;
    if (cdst) cdst[c] = (bf16_t)v;
  }
}

// Attention: per (call,batch,head) 37x37 scores, softmax, @V (all small: VALU)
__constant__ int d_qc[24] = {2,3, 3,0, 8,1, 6,7, 7,4, 9,5, 7,2, 3,6, 7,3, 7,8, 9,3, 8,9};
__constant__ int d_kc[24] = {3,2, 0,3, 1,8, 7,6, 4,7, 5,9, 2,7, 6,3, 3,7, 8,7, 3,9, 9,8};

__global__ __launch_bounds__(256)
void k_attention(const bf16_t* qkv, bf16_t* attn) {
  int blk = blockIdx.x;
  int call = blk >> 6, rem = blk & 63, b = rem >> 2, h = rem & 3;
  int qch = d_qc[call], kch = d_kc[call];
  const bf16_t* Qb = qkv + ((long)(qch * 16 + b) * 37) * 6144 + h * 512;
  const bf16_t* Kb = qkv + ((long)(kch * 16 + b) * 37) * 6144 + 2048 + h * 512;
  const bf16_t* Vb = qkv + ((long)(kch * 16 + b) * 37) * 6144 + 4096 + h * 512;
  __shared__ float S[37 * 37];
  const float scale = 0.04419417382415922f;  // 1/sqrt(512)
  for (int idx = threadIdx.x; idx < 37 * 37; idx += 256) {
    int qi = idx / 37, ki = idx % 37;
    const bf16_t* q = Qb + (long)qi * 6144;
    const bf16_t* k = Kb + (long)ki * 6144;
    float s = 0.0f;
    for (int d = 0; d < 512; ++d) s += (float)q[d] * (float)k[d];
    S[idx] = s * scale;
  }
  __syncthreads();
  if (threadIdx.x < 37) {
    float* row = S + threadIdx.x * 37;
    float m = row[0];
    for (int i = 1; i < 37; ++i) m = fmaxf(m, row[i]);
    float sum = 0.0f;
    for (int i = 0; i < 37; ++i) { float e = __expf(row[i] - m); row[i] = e; sum += e; }
    float inv = 1.0f / sum;
    for (int i = 0; i < 37; ++i) row[i] *= inv;
  }
  __syncthreads();
  int rh = call >> 1, half = call & 1;   // call even = ao1 (rows 0..36)
  bf16_t* Ob = attn + ((long)(rh * 16 + b) * 74 + half * 37) * 2048 + h * 512;
  for (int idx = threadIdx.x; idx < 37 * 512; idx += 256) {
    int qi = idx / 512, d = idx % 512;
    float s = 0.0f;
    for (int ki = 0; ki < 37; ++ki) s += S[qi * 37 + ki] * (float)Vb[(long)ki * 6144 + d];
    Ob[(long)qi * 2048 + d] = (bf16_t)s;
  }
}

// l1 (time-74 -> 10) + leaky + flatten(c-major) + l2 + sigmoid
__constant__ int d_rh_pos[12] = {1, 2, 0, 4, 5, 3, 11, 10, 9, 8, 7, 6};

__global__ void k_head(const float* z2, const float* l1w, const float* l1b,
                       const float* l2w, const float* l2b, float* out) {
  int tid = blockIdx.x * blockDim.x + threadIdx.x;
  if (tid >= 192) return;
  int rh = tid >> 4, b = tid & 15;
  const float* z = z2 + ((long)(rh * 16 + b) * 74) * 32;
  float acc = 0.0f;
  for (int c = 0; c < 32; ++c)
    for (int j = 0; j < 10; ++j) {
      float s = l1b[j];
      for (int t = 0; t < 74; ++t) s += z[t * 32 + c] * l1w[j * 74 + t];
      s = s >= 0.0f ? s : 0.2f * s;
      acc += s * l2w[c * 10 + j];
    }
  acc += l2b[0];
  out[(long)d_rh_pos[rh] * 16 + b] = 1.0f / (1.0f + __expf(-acc));
}

__global__ void k_vsoftmax(const float* mean, float* out) {
  int b = blockIdx.x;
  if (threadIdx.x == 0) {
    float v[20];
    float m = -1e30f;
    for (int i = 0; i < 20; ++i) { v[i] = mean[(long)b * 20 + i]; m = fmaxf(m, v[i]); }
    float s = 0.0f;
    for (int i = 0; i < 20; ++i) { v[i] = __expf(v[i] - m); s += v[i]; }
    for (int i = 0; i < 20; ++i) out[(long)b * 20 + i] = v[i] / s;
  }
}

// ---------------------------------------------------------------------------
// Host orchestration
// ---------------------------------------------------------------------------
static inline void gemm(hipStream_t st, const bf16_t* A, long arpb, long abst, long lda,
                        const bf16_t* B, const float* bias, float* outF, bf16_t* outB,
                        long orpb, long obst, long old_, int M, int N, int K, int act) {
  GemmP p{A, arpb, abst, lda, B, bias, outF, outB, orpb, obst, old_, M, N, K, act};
  dim3 g((N + BLKT - 1) / BLKT, (M + BLKT - 1) / BLKT);
  k_gemm_wmma<<<g, 256, 0, st>>>(p);
}

extern "C" void kernel_launch(void* const* d_in, const int* in_sizes, int n_in,
                              void* d_out_v, int out_size, void* d_ws, size_t ws_size,
                              hipStream_t stream) {
  (void)in_sizes; (void)n_in; (void)out_size; (void)ws_size;
  const float* x      = (const float*)d_in[0];
  const float* w_emb  = (const float*)d_in[1];
  const float* b_emb  = (const float*)d_in[2];
  const float* w_cls  = (const float*)d_in[3];
  const float* tf1_w  = (const float*)d_in[4];
  const float* tf1_b  = (const float*)d_in[5];
  const float* tf2_w  = (const float*)d_in[6];
  const float* tf2_b  = (const float*)d_in[7];
  const float* in_w   = (const float*)d_in[8];
  const float* in_b   = (const float*)d_in[9];
  const float* out_w  = (const float*)d_in[10];
  const float* out_b  = (const float*)d_in[11];
  const float* c1_w   = (const float*)d_in[12];
  const float* c1_b   = (const float*)d_in[13];
  const float* c2_w   = (const float*)d_in[14];
  const float* c2_b   = (const float*)d_in[15];
  const float* l1_w   = (const float*)d_in[16];
  const float* l1_b   = (const float*)d_in[17];
  const float* l2_w   = (const float*)d_in[18];
  const float* l2_b   = (const float*)d_in[19];
  float* d_out = (float*)d_out_v;

  // output layout (flat, return order)
  const long OFF_VS = 0, OFF_ACT = 320, OFF_A2 = 12320, OFF_CAS = 24320;
  const long OFF_RH = 264320, OFF_EA = 264512, OFF_EB = 5179712;
  const long OFF_HA = 10094912, OFF_HB = 11307328;

  // workspace carve-up (~480 MB)
  char* wsb = (char*)d_ws;
  size_t off = 0;
  auto alloc = [&](size_t bytes) { size_t o = off; off = (off + bytes + 255) & ~(size_t)255; return o; };
  bf16_t* xpad   = (bf16_t*)(wsb + alloc((size_t)NB * (NT + 2) * NC * 2));
  bf16_t* Wemb   = (bf16_t*)(wsb + alloc((size_t)3 * NC * NC * 2));
  bf16_t* Wcls   = (bf16_t*)(wsb + alloc((size_t)NC * 20 * 2));
  bf16_t* Win    = (bf16_t*)(wsb + alloc((size_t)NC * 6144 * 2));
  bf16_t* Wout   = (bf16_t*)(wsb + alloc((size_t)NC * NC * 2));
  bf16_t* Wc1    = (bf16_t*)(wsb + alloc((size_t)3 * NC * 128 * 2));
  bf16_t* Wc2    = (bf16_t*)(wsb + alloc((size_t)3 * 128 * 32 * 2));
  float*  embF   = (float*)(wsb + alloc((size_t)NB * NT * NC * 4));
  bf16_t* embB   = (bf16_t*)(wsb + alloc((size_t)NB * NT * NC * 2));
  bf16_t* qkv    = (bf16_t*)(wsb + alloc((size_t)10 * 16 * 37 * 6144 * 2));
  bf16_t* chunks = (bf16_t*)(wsb + alloc((size_t)10 * 16 * 37 * 2048 * 2));
  bf16_t* attn   = (bf16_t*)(wsb + alloc((size_t)12 * 16 * 74 * 2048 * 2));
  bf16_t* ypad   = (bf16_t*)(wsb + alloc((size_t)12 * 16 * 76 * 2048 * 2));
  bf16_t* z1pad  = (bf16_t*)(wsb + alloc((size_t)12 * 16 * 76 * 128 * 2));
  float*  z2     = (float*)(wsb + alloc((size_t)12 * 16 * 74 * 32 * 4));
  float*  rev    = (float*)(wsb + alloc((size_t)NB * NT * 4));
  float*  hsa    = (float*)(wsb + alloc((size_t)NB * NT * 4));
  float*  hsb    = (float*)(wsb + alloc((size_t)NB * NT * 4));
  int*    idxEA  = (int*)(wsb + alloc((size_t)NB * 150 * 4));
  int*    idxEB  = (int*)(wsb + alloc((size_t)NB * 150 * 4));
  int*    idxHA  = (int*)(wsb + alloc((size_t)NB * 37 * 4));
  int*    idxHB  = (int*)(wsb + alloc((size_t)NB * 37 * 4));
  float*  median = (float*)(wsb + alloc((size_t)NB * 4));
  float*  vsmean = (float*)(wsb + alloc((size_t)NB * 20 * 4));

  // zero padded buffers (pad rows must be 0 every call; ws is not re-zeroed)
  k_fill_zero<<<2048, 256, 0, stream>>>((unsigned*)xpad,  (long)NB * (NT + 2) * NC / 2);
  k_fill_zero<<<2048, 256, 0, stream>>>((unsigned*)ypad,  (long)12 * 16 * 76 * 2048 / 2);
  k_fill_zero<<<2048, 256, 0, stream>>>((unsigned*)z1pad, (long)12 * 16 * 76 * 128 / 2);

  // bf16 conversions / weight re-layout to [KW][Ci][N]
  k_x_to_xpad<<<4096, 256, 0, stream>>>(x, xpad);
  k_w_to_bf16<<<4096, 256, 0, stream>>>(w_emb, Wemb, 2048, 2048, 3);
  k_w_to_bf16<<<512,  256, 0, stream>>>(w_cls, Wcls, 20, 2048, 1);
  k_w_to_bf16<<<4096, 256, 0, stream>>>(in_w,  Win, 6144, 2048, 1);
  k_w_to_bf16<<<4096, 256, 0, stream>>>(out_w, Wout, 2048, 2048, 1);
  k_w_to_bf16<<<1024, 256, 0, stream>>>(c1_w,  Wc1, 128, 2048, 3);
  k_w_to_bf16<<<64,   256, 0, stream>>>(c2_w,  Wc2, 32, 128, 3);

  // embed conv as GEMM over padded windows: M=12000, K=6144, N=2048, relu
  gemm(stream, xpad, NT, (long)(NT + 2) * NC, NC, Wemb, b_emb, embF, embB,
       NB * NT, 0, NC, NB * NT, NC, 3 * NC, 1);
  // cas: M=12000, K=2048, N=20, relu -> straight into d_out
  gemm(stream, embB, NB * NT, 0, NC, Wcls, nullptr, d_out + OFF_CAS, nullptr,
       NB * NT, 0, 20, NB * NT, 20, NC, 1);

  k_actionness<<<(NB * NT + 255) / 256, 256, 0, stream>>>(d_out + OFF_CAS, d_out + OFF_ACT);
  k_a2<<<NB, 256, 0, stream>>>(d_out + OFF_ACT, tf1_w, tf1_b, tf2_w, tf2_b, d_out + OFF_A2);

  // selections
  k_sort_select<<<NB, 256, 0, stream>>>(d_out + OFF_ACT, NT, 0, 1, 1, NT,
                                        idxEA, 150, median, nullptr, 0);
  k_rev<<<NB, 256, 0, stream>>>(d_out + OFF_ACT, rev);
  k_sort_select<<<NB, 256, 0, stream>>>(rev, NT, 0, 1, 1, NT, idxEB, 150, nullptr, nullptr, 0);
  k_morph<<<NB, 256, 0, stream>>>(d_out + OFF_ACT, median, hsa, hsb);
  k_sort_select<<<NB, 256, 0, stream>>>(hsa, NT, 0, 1, 1, NT, idxHA, 37, nullptr, nullptr, 0);
  k_sort_select<<<NB, 256, 0, stream>>>(hsb, NT, 0, 1, 1, NT, idxHB, 37, nullptr, nullptr, 0);

  // gathers -> outputs + bf16 chunk buffers (chunks 0-3 ea, 4-7 eb, 8 ha, 9 hb)
  k_gather<<<NB * 150, 256, 0, stream>>>(embF, idxEA, 150, d_out + OFF_EA, chunks, 0, 148);
  k_gather<<<NB * 150, 256, 0, stream>>>(embF, idxEB, 150, d_out + OFF_EB, chunks, 4, 148);
  k_gather<<<NB * 37,  256, 0, stream>>>(embF, idxHA, 37,  d_out + OFF_HA, chunks, 8, 37);
  k_gather<<<NB * 37,  256, 0, stream>>>(embF, idxHB, 37,  d_out + OFF_HB, chunks, 9, 37);

  // hoisted QKV for all 10 chunks: M=5920, K=2048, N=6144
  gemm(stream, chunks, 5920, 0, NC, Win, in_b, nullptr, qkv, 5920, 0, 6144, 5920, 6144, NC, 0);

  // attention (24 calls x 16 batch x 4 heads)
  k_attention<<<24 * 64, 256, 0, stream>>>(qkv, attn);

  // batched out-projection into padded y: M=14208, K=2048, N=2048
  gemm(stream, attn, 14208, 0, NC, Wout, out_b, nullptr, ypad + 2048,
       74, (long)76 * 2048, 2048, 14208, 2048, NC, 0);
  // conv1 as GEMM: K=3*2048, N=128, leaky
  gemm(stream, ypad, 74, (long)76 * 2048, 2048, Wc1, c1_b, nullptr, z1pad + 128,
       74, (long)76 * 128, 128, 14208, 128, 6144, 2);
  // conv2 as GEMM: K=3*128, N=32, leaky
  gemm(stream, z1pad, 74, (long)76 * 128, 128, Wc2, c2_b, z2, nullptr,
       14208, 0, 32, 14208, 32, 384, 2);

  // l1/l2/sigmoid heads -> 12 scalar outputs in return order
  k_head<<<1, 192, 0, stream>>>(z2, l1_w, l1_b, l2_w, l2_b, d_out + OFF_RH);

  // video scores: per-(b,cls) mean of top-150 cas values, then softmax over 20
  k_sort_select<<<NB * 20, 256, 0, stream>>>(d_out + OFF_CAS, (long)NT * 20, 1, 20, 20, NT,
                                             nullptr, 0, nullptr, vsmean, 150);
  k_vsoftmax<<<NB, 32, 0, stream>>>(vsmean, d_out + OFF_VS);
}
